// TransformerMemoryUpdater_31224412242762
// MI455X (gfx1250) — matrix operand order repaired and checked by reading, not compiled, not run on hardware
//
#include <hip/hip_runtime.h>
#include <hip/hip_bf16.h>

// ---------------------------------------------------------------------------
// Problem constants (from reference)
// ---------------------------------------------------------------------------
#define NN_   131072
#define MM_   8
#define DIN_  128
#define DT_   64
#define DKV_  192      // DIN + DT
#define DOUT_ 128
#define HH_   4
#define DH_   32
#define LN_EPS_ 1e-5f

// WMMA vector types (CDNA5 wave32, 16x16x32 bf16 -> f32)
typedef __attribute__((ext_vector_type(16))) __bf16 v16bf;
typedef __attribute__((ext_vector_type(8)))  __bf16 bf16x8;
typedef __attribute__((ext_vector_type(8)))  float  v8f;

// ---------------------------------------------------------------------------
// WMMA layout used here (CDNA5 16x16x32 bf16; A/B k order is this file's contiguous choice, same in A and B):
// A (16x32): lane L<16 -> row L,    K = {k0..k0+15}
//            lane L>=16 -> row L-16, K = {k0+16..k0+31}
// B (32x16): lane L<16 -> col L,    K = {k0..k0+15}
//            lane L>=16 -> col L-16, K = {k0+16..k0+31}
// C/D (16x16 f32): VGPR r, lane L -> row r + 8*(L>=16), col = L&15
// ---------------------------------------------------------------------------
__device__ inline v16bf ld_a(const __bf16* base, int stride, int row0, int k0, int lane) {
    int row = row0 + (lane & 15);
    int kb  = k0 + ((lane & 16) ? 16 : 0);
    const __bf16* p = base + row * stride + kb;
    bf16x8 lo = *(const bf16x8*)(p);
    bf16x8 hi = *(const bf16x8*)(p + 8);
    v16bf a;
#pragma unroll
    for (int i = 0; i < 8; ++i) { a[i] = lo[i]; a[i + 8] = hi[i]; }
    return a;
}

// B tiles are pre-packed by the prep kernel: tile t occupies 512 contiguous
// bf16 (1 KB); within a tile, lane L's 16 elements are at [L*16 .. L*16+15].
// A wave-level B-tile load is therefore one fully coalesced contiguous 1 KB read.
__device__ inline v16bf ld_bt(const __bf16* wt, int tile, int lane) {
    const __bf16* p = wt + tile * 512 + lane * 16;
    bf16x8 lo = *(const bf16x8*)(p);
    bf16x8 hi = *(const bf16x8*)(p + 8);
    v16bf b;
#pragma unroll
    for (int i = 0; i < 8; ++i) { b[i] = lo[i]; b[i + 8] = hi[i]; }
    return b;
}

__device__ inline v8f wmma_bf16(v16bf a, v16bf b, v8f c) {
    return __builtin_amdgcn_wmma_f32_16x16x32_bf16(false, a, false, b, (short)0, c,
                                                   false, false);
}

// ---------------------------------------------------------------------------
// Prep: fp32 weights -> bf16, repacked into WMMA B-tile layout.
// For a weight W[DOUT x S] (row = output col j, inner = k):
//   tiles indexed t = j_tile * (S/32) + k_chunk, 512 elems each.
//   elem e in tile: lane = e>>4, i = e&15,
//     col = j_tile*16 + (lane&15), k = k_chunk*32 + ((lane&16)?16:0) + i.
// ws layout (bf16 elems): wq[16384] | wk[24576] | wv[24576] | mlp[16384]
// ---------------------------------------------------------------------------
__device__ inline __bf16 pack_elem(const float* __restrict__ W, int S, int CT, int idx) {
    int tile = idx >> 9;
    int e    = idx & 511;
    int lane = e >> 4;
    int i    = e & 15;
    int j    = tile / CT;
    int c    = tile - j * CT;
    int col  = j * 16 + (lane & 15);
    int k    = c * 32 + ((lane & 16) ? 16 : 0) + i;
    return (__bf16)W[col * S + k];
}

__global__ void tmu_prep_kernel(const float* __restrict__ wq,
                                const float* __restrict__ wk,
                                const float* __restrict__ wv,
                                const float* __restrict__ mlp,
                                __bf16* __restrict__ ws) {
    int i = blockIdx.x * 256 + threadIdx.x;
    if (i < 16384)       ws[i] = pack_elem(wq,  DOUT_, 4, i);
    else if (i < 40960)  ws[i] = pack_elem(wk,  DKV_,  6, i - 16384);
    else if (i < 65536)  ws[i] = pack_elem(wv,  DKV_,  6, i - 40960);
    else if (i < 81920)  ws[i] = pack_elem(mlp, DOUT_, 4, i - 65536);
}

// ---------------------------------------------------------------------------
// Main fused kernel. 256 threads (8 waves), 16 nodes per block.
// ---------------------------------------------------------------------------
__global__ __launch_bounds__(256, 1)
void tmu_main_kernel(const float* __restrict__ mem,
                     const float* __restrict__ mem_input,
                     const float* __restrict__ ts,
                     const float* __restrict__ mail_ts,
                     const float* __restrict__ wq_b,
                     const float* __restrict__ wk_b,
                     const float* __restrict__ wv_b,
                     const float* __restrict__ ln_g,
                     const float* __restrict__ ln_b,
                     const float* __restrict__ mlp_b,
                     const float* __restrict__ time_w,
                     const float* __restrict__ time_b,
                     const __bf16* __restrict__ wq_t,
                     const __bf16* __restrict__ wk_t,
                     const __bf16* __restrict__ wv_t,
                     const __bf16* __restrict__ mlp_t,
                     float* __restrict__ out) {
    __shared__ __bf16 s_mails[128 * DKV_];   // 49152 B: 16 nodes x 8 slots x 192
    __shared__ __bf16 s_memb[16 * DOUT_];    //  4096 B
    __shared__ float  s_q[16 * DOUT_];       //  8192 B
    __shared__ __bf16 s_k[128 * DOUT_];      // 32768 B
    __shared__ __bf16 s_v[128 * DOUT_];      // 32768 B
    __shared__ float  s_att[16 * MM_ * HH_]; //  2048 B
    __shared__ __bf16 s_rln[16 * DOUT_];     //  4096 B

    const int tid  = threadIdx.x;
    const int wave = tid >> 5;
    const int lane = tid & 31;
    const int nb   = blockIdx.x * 16;        // first node of this block

    // ---- Phase A: stage activations into LDS as bf16 -----------------------
#pragma unroll
    for (int i = 0; i < 8; ++i) {            // mem: 2048 f32
        int idx = tid + 256 * i;
        s_memb[idx] = (__bf16)mem[nb * DOUT_ + idx];
    }
#pragma unroll 4
    for (int i = 0; i < 64; ++i) {           // mem_input: 16384 f32
        int idx = tid + 256 * i;
        int row = idx >> 7, col = idx & 127;
        s_mails[row * DKV_ + col] = (__bf16)mem_input[nb * (MM_ * DIN_) + idx];
    }
    {                                        // time encoding: cos(dt*w + b)
        int row  = tid >> 1;                 // mail row in [0,128)
        int t0   = (tid & 1) * 32;
        int node = nb + (row >> 3);
        int m    = row & 7;
        float dt = ts[node] - mail_ts[node * MM_ + m];
#pragma unroll
        for (int i = 0; i < 32; ++i) {
            int t = t0 + i;
            float f = __cosf(dt * time_w[t] + time_b[t]);
            s_mails[row * DKV_ + DIN_ + t] = (__bf16)f;
        }
    }
    __syncthreads();

    // ---- Phase B1: Q = mem @ WqT + b (wave w -> col tile 16w) --------------
    {
        v8f acc = {};
#pragma unroll
        for (int c = 0; c < 4; ++c) {
            v16bf a = ld_a(s_memb, DOUT_, 0, c * 32, lane);
            v16bf b = ld_bt(wq_t, wave * 4 + c, lane);
            acc = wmma_bf16(a, b, acc);
        }
        int j0  = wave * 16;
        int col = lane & 15;
        int rb  = (lane & 16) ? 8 : 0;
        float bias = wq_b[j0 + col];
#pragma unroll
        for (int r = 0; r < 8; ++r)
            s_q[(r + rb) * DOUT_ + j0 + col] = acc[r] + bias;
    }

    // ---- Phase B2: K,V = mails @ W{k,v}T + b (wave w -> rows 16w..16w+15) --
    {
        const int row0 = wave * 16;
#pragma unroll
        for (int j = 0; j < 8; ++j) {
            v8f ak = {}, av = {};
#pragma unroll
            for (int c = 0; c < 6; ++c) {
                v16bf a  = ld_a(s_mails, DKV_, row0, c * 32, lane);
                v16bf bk = ld_bt(wk_t, j * 6 + c, lane);
                ak = wmma_bf16(a, bk, ak);
                v16bf bv = ld_bt(wv_t, j * 6 + c, lane);
                av = wmma_bf16(a, bv, av);
            }
            int j0  = j * 16;
            int col = lane & 15;
            int rb  = (lane & 16) ? 8 : 0;
            float bkk = wk_b[j0 + col];
            float bvv = wv_b[j0 + col];
#pragma unroll
            for (int r = 0; r < 8; ++r) {
                s_k[(row0 + r + rb) * DOUT_ + j0 + col] = (__bf16)(ak[r] + bkk);
                s_v[(row0 + r + rb) * DOUT_ + j0 + col] = (__bf16)(av[r] + bvv);
            }
        }
    }
    __syncthreads();

    // ---- Phase C: attention scores + LeakyReLU + softmax over M ------------
    {
        int nn   = lane >> 4;                // node within wave's pair
        int rem  = lane & 15;
        int m    = rem & 7;
        int hp   = rem >> 3;                 // head parity
        int ln16 = wave * 2 + nn;            // local node in [0,16)
        int krow = ln16 * MM_ + m;
#pragma unroll
        for (int hi = 0; hi < 2; ++hi) {
            int h = hp + 2 * hi;
            float s = 0.f;
#pragma unroll
            for (int d = 0; d < DH_; ++d)
                s += s_q[ln16 * DOUT_ + h * DH_ + d] *
                     (float)s_k[krow * DOUT_ + h * DH_ + d];
            s = (s >= 0.f) ? s : 0.2f * s;   // LeakyReLU(0.2)
            float mx = s;                     // softmax over 8 slot-lanes
            for (int msk = 1; msk <= 4; msk <<= 1)
                mx = fmaxf(mx, __shfl_xor(mx, msk, 32));
            float e = __expf(s - mx);
            float tot = e;
            for (int msk = 1; msk <= 4; msk <<= 1)
                tot += __shfl_xor(tot, msk, 32);
            s_att[ln16 * 32 + m * 4 + h] = e / tot;
        }
    }
    __syncthreads();

    // ---- Phase D: out = att@V + mem, LayerNorm -----------------------------
    {
        int nn    = lane >> 4;
        int rem   = lane & 15;
        int ln16  = wave * 2 + nn;
        int gnode = nb + ln16;
        float vals[8];
        float psum = 0.f;
#pragma unroll
        for (int i = 0; i < 8; ++i) {
            int j = i * 16 + rem;
            int h = j >> 5;
            float acc = 0.f;
#pragma unroll
            for (int m = 0; m < MM_; ++m)
                acc += s_att[ln16 * 32 + m * 4 + h] *
                       (float)s_v[(ln16 * MM_ + m) * DOUT_ + j];
            acc += mem[gnode * DOUT_ + j];   // residual
            vals[i] = acc;
            psum += acc;
        }
        for (int msk = 1; msk <= 8; msk <<= 1)   // reduce across 16 lanes/node
            psum += __shfl_xor(psum, msk, 32);
        float mu = psum * (1.f / 128.f);
        float pvar = 0.f;
#pragma unroll
        for (int i = 0; i < 8; ++i) { float d = vals[i] - mu; pvar += d * d; }
        for (int msk = 1; msk <= 8; msk <<= 1)
            pvar += __shfl_xor(pvar, msk, 32);
        float rstd = __frsqrt_rn(pvar * (1.f / 128.f) + LN_EPS_);
#pragma unroll
        for (int i = 0; i < 8; ++i) {
            int j = i * 16 + rem;
            float r = (vals[i] - mu) * rstd * ln_g[j] + ln_b[j];
            s_rln[ln16 * DOUT_ + j] = (__bf16)r;
        }
    }
    __syncthreads();

    // ---- Phase E: MLP + ReLU + store (wave w -> col tile 16w) --------------
    {
        v8f acc = {};
#pragma unroll
        for (int c = 0; c < 4; ++c) {
            v16bf a = ld_a(s_rln, DOUT_, 0, c * 32, lane);
            v16bf b = ld_bt(mlp_t, wave * 4 + c, lane);
            acc = wmma_bf16(a, b, acc);
        }
        int j0  = wave * 16;
        int col = lane & 15;
        int rb  = (lane & 16) ? 8 : 0;
        float bias = mlp_b[j0 + col];
#pragma unroll
        for (int r = 0; r < 8; ++r) {
            float v = acc[r] + bias;
            out[(nb + r + rb) * DOUT_ + j0 + col] = fmaxf(v, 0.f);
        }
    }
}

// ---------------------------------------------------------------------------
// Launcher
// ---------------------------------------------------------------------------
extern "C" void kernel_launch(void* const* d_in, const int* in_sizes, int n_in,
                              void* d_out, int out_size, void* d_ws, size_t ws_size,
                              hipStream_t stream) {
    const float* mem       = (const float*)d_in[0];
    const float* mem_input = (const float*)d_in[1];
    const float* ts        = (const float*)d_in[2];
    const float* mail_ts   = (const float*)d_in[3];
    const float* wq_w      = (const float*)d_in[4];
    const float* wq_b      = (const float*)d_in[5];
    const float* wk_w      = (const float*)d_in[6];
    const float* wk_b      = (const float*)d_in[7];
    const float* wv_w      = (const float*)d_in[8];
    const float* wv_b      = (const float*)d_in[9];
    const float* ln_g      = (const float*)d_in[10];
    const float* ln_b      = (const float*)d_in[11];
    const float* mlp_w     = (const float*)d_in[12];
    const float* mlp_b     = (const float*)d_in[13];
    const float* time_w    = (const float*)d_in[14];
    const float* time_b    = (const float*)d_in[15];

    __bf16* wsb    = (__bf16*)d_ws;
    __bf16* wq_t   = wsb;
    __bf16* wk_t   = wsb + 16384;
    __bf16* wv_t   = wsb + 40960;
    __bf16* mlp_t  = wsb + 65536;

    const int n = in_sizes[2];               // number of nodes (ts has N elems)

    tmu_prep_kernel<<<320, 256, 0, stream>>>(wq_w, wk_w, wv_w, mlp_w, wsb);

    tmu_main_kernel<<<n / 16, 256, 0, stream>>>(
        mem, mem_input, ts, mail_ts,
        wq_b, wk_b, wv_b, ln_g, ln_b, mlp_b, time_w, time_b,
        wq_t, wk_t, wv_t, mlp_t,
        (float*)d_out);
}
